// Block_7627861918171
// MI455X (gfx1250) — matrix operand order repaired and checked by reading, not compiled, not run on hardware
//
#include <hip/hip_runtime.h>
#include <hip/hip_bf16.h>
#include <math.h>

#define DEV __device__ __forceinline__

typedef __attribute__((ext_vector_type(16))) _Float16 v16h;
typedef __attribute__((ext_vector_type(8)))  float    v8f;

namespace {

constexpr int Bv   = 2;
constexpr int NTOK = 2240;
constexpr int Cv   = 384;
constexpr int Hh   = 6;
constexpr int HD   = 64;
constexpr int Mv   = Bv * NTOK;   // 4480 rows of the residual stream
constexpr int TC   = 3 * Cv;      // 1152
constexpr int HID  = 1536;

union Frag {
    v16h     v;
    float4   q2[2];
    _Float16 h[16];
};

// A-fragment 16x32 f16 (MxK). Per ISA: lane L holds row M=L%16;
// halfs[0..7] -> K = 8*(L/16)+j, halfs[8..15] -> K = 16+8*(L/16)+j.
DEV v16h load_a(const _Float16* base, int row, int ld, int k0, int lane) {
    const _Float16* p = base + (size_t)row * ld + k0 + ((lane >> 4) << 3);
    Frag f;
    f.q2[0] = *(const float4*)(p);
    f.q2[1] = *(const float4*)(p + 16);
    return f.v;
}

// Same fragment but converting from an f32 source (attn probabilities).
DEV v16h load_a_f32(const float* base, int row, int ld, int k0, int lane) {
    const float* p = base + (size_t)row * ld + k0 + ((lane >> 4) << 3);
    Frag f;
#pragma unroll
    for (int j = 0; j < 8; ++j) f.h[j] = (_Float16)p[j];
#pragma unroll
    for (int j = 0; j < 8; ++j) f.h[8 + j] = (_Float16)p[16 + j];
    return f.v;
}

// B-fragment 32x16 f16 (KxN). Lane L holds column N=L%16;
// halfs[0..7] -> K = 8*(L/16)+j, halfs[8..15] -> K = 16+8*(L/16)+j. NT source: column n == row n of W.
DEV v16h load_b(const _Float16* base, int row, int ld, int k0, int lane) {
    const _Float16* p = base + (size_t)row * ld + k0 + ((lane >> 4) << 3);
    Frag f;
    f.q2[0] = *(const float4*)(p);
    f.q2[1] = *(const float4*)(p + 16);
    return f.v;
}

DEV v8f wmma_f16(v16h a, v16h b, v8f c) {
    return __builtin_amdgcn_wmma_f32_16x16x32_f16(false, a, false, b, (short)0, c,
                                                  false, false);
}

// One wave computes a 16(M) x 64(N) tile: A fragment reused across 4 B fragments.
DEV void gemm_nt_16x64(const _Float16* A, int lda, const _Float16* W, int ldw,
                       int m0, int n0, int K, int lane, v8f acc[4]) {
    const int ar = m0 + (lane & 15);
    for (int k0 = 0; k0 < K; k0 += 32) {
        v16h a = load_a(A, ar, lda, k0, lane);
#pragma unroll
        for (int s = 0; s < 4; ++s) {
            v16h b = load_b(W, n0 + s * 16 + (lane & 15), ldw, k0, lane);
            acc[s] = wmma_f16(a, b, acc[s]);
        }
    }
}

} // namespace

// ---------------------------------------------------------------- utilities

__global__ void cvt_f32_to_f16_kernel(const float* __restrict__ src,
                                      _Float16* __restrict__ dst, int n) {
    int i = blockIdx.x * blockDim.x + threadIdx.x;
    if (i < n) dst[i] = (_Float16)src[i];
}

// LayerNorm over C=384. One 128-thread block per row (3 cols/thread).
// Optionally adds positional embedding (rep = N/N_TOKEN = 1) and stores the
// f32 residual stream; always emits the f16 normalized activations.
__global__ void ln_kernel(const float* __restrict__ xin,
                          const float* __restrict__ pos,
                          const float* __restrict__ g,
                          const float* __restrict__ beta,
                          float* __restrict__ xout,
                          _Float16* __restrict__ hout) {
    __shared__ float red[128];
    const int row = blockIdx.x;
    const int tok = row % NTOK;
    const int t = threadIdx.x;
    float v[3];
    float s = 0.f;
#pragma unroll
    for (int j = 0; j < 3; ++j) {
        int c = t + 128 * j;
        float val = xin[(size_t)row * Cv + c];
        if (pos) val += pos[(size_t)tok * Cv + c];
        v[j] = val;
        s += val;
    }
    red[t] = s; __syncthreads();
    for (int o = 64; o > 0; o >>= 1) { if (t < o) red[t] += red[t + o]; __syncthreads(); }
    const float mu = red[0] * (1.f / Cv);
    __syncthreads();
    float s2 = 0.f;
#pragma unroll
    for (int j = 0; j < 3; ++j) { float d = v[j] - mu; s2 += d * d; }
    red[t] = s2; __syncthreads();
    for (int o = 64; o > 0; o >>= 1) { if (t < o) red[t] += red[t + o]; __syncthreads(); }
    const float rstd = rsqrtf(red[0] * (1.f / Cv) + 1e-5f);
#pragma unroll
    for (int j = 0; j < 3; ++j) {
        int c = t + 128 * j;
        if (xout) xout[(size_t)row * Cv + c] = v[j];
        hout[(size_t)row * Cv + c] = (_Float16)((v[j] - mu) * rstd * g[c] + beta[c]);
    }
}

// qkv = ln1(x) @ w_qkv^T ; epilogue scatters into q (pre-scaled), k [B,H,N,hd]
// and v transposed [B,H,hd,N] f16 layouts for the attention GEMMs.
__global__ void gemm_qkv_kernel(const _Float16* __restrict__ h16,
                                const _Float16* __restrict__ w16,
                                _Float16* __restrict__ q16,
                                _Float16* __restrict__ k16,
                                _Float16* __restrict__ vT16) {
    const int lane = threadIdx.x & 31;
    const int wave = (blockIdx.x * blockDim.x + threadIdx.x) >> 5;
    constexpr int TN = TC / 64;           // 18
    constexpr int TM = Mv / 16;           // 280
    if (wave >= TM * TN) return;
    const int m0 = (wave / TN) * 16;
    const int n0 = (wave % TN) * 64;
    v8f acc[4] = {};
    gemm_nt_16x64(h16, Cv, w16, Cv, m0, n0, Cv, lane, acc);
    const int col = lane & 15;
    const int rbase = (lane >> 4) << 3;
    constexpr float scale = 0.125f;       // hd^-0.5, hd = 64
#pragma unroll
    for (int s = 0; s < 4; ++s) {
        const int n = n0 + s * 16 + col;
        const int sec = n / Cv, within = n % Cv;
        const int hd = within >> 6, d = within & 63;
#pragma unroll
        for (int r = 0; r < 8; ++r) {
            const int m = m0 + rbase + r;
            const int b = m / NTOK, tok = m % NTOK;
            const float val = acc[s][r];
            if (sec == 0)
                q16[(((size_t)(b * Hh + hd) * NTOK) + tok) * HD + d] = (_Float16)(val * scale);
            else if (sec == 1)
                k16[(((size_t)(b * Hh + hd) * NTOK) + tok) * HD + d] = (_Float16)val;
            else
                vT16[(((size_t)(b * Hh + hd) * HD) + d) * NTOK + tok] = (_Float16)val;
        }
    }
}

// scores[b,h] = (q*scale) @ k^T, written f32 straight into the attn output.
__global__ void gemm_scores_kernel(const _Float16* __restrict__ q16,
                                   const _Float16* __restrict__ k16,
                                   float* __restrict__ attn) {
    const int lane = threadIdx.x & 31;
    const int wave = (blockIdx.x * blockDim.x + threadIdx.x) >> 5;
    constexpr int TM = NTOK / 16;         // 140
    constexpr int TN = NTOK / 64;         // 35
    constexpr int PER = TM * TN;          // 4900 per (b,h)
    if (wave >= PER * Bv * Hh) return;
    const int bh = wave / PER;
    const int t  = wave % PER;
    const int m0 = (t / TN) * 16;
    const int n0 = (t % TN) * 64;
    const _Float16* Q = q16 + (size_t)bh * NTOK * HD;
    const _Float16* K = k16 + (size_t)bh * NTOK * HD;
    float* S = attn + (size_t)bh * NTOK * NTOK;
    v8f acc[4] = {};
    gemm_nt_16x64(Q, HD, K, HD, m0, n0, HD, lane, acc);
    const int col = lane & 15;
    const int rbase = (lane >> 4) << 3;
#pragma unroll
    for (int s = 0; s < 4; ++s)
#pragma unroll
        for (int r = 0; r < 8; ++r)
            S[(size_t)(m0 + rbase + r) * NTOK + n0 + s * 16 + col] = acc[s][r];
}

// In-place softmax over the last axis of attn. One 256-thread block per row.
__global__ void softmax_kernel(float* __restrict__ attn) {
    __shared__ float red[256];
    float* p = attn + (size_t)blockIdx.x * NTOK;
    const int t = threadIdx.x;
    float vals[9];
    float mx = -3.4e38f;
    int cnt = 0;
    for (int c = t; c < NTOK; c += 256, ++cnt) { vals[cnt] = p[c]; mx = fmaxf(mx, vals[cnt]); }
    red[t] = mx; __syncthreads();
    for (int o = 128; o > 0; o >>= 1) { if (t < o) red[t] = fmaxf(red[t], red[t + o]); __syncthreads(); }
    mx = red[0]; __syncthreads();
    float sum = 0.f;
    for (int i = 0; i < cnt; ++i) { vals[i] = expf(vals[i] - mx); sum += vals[i]; }
    red[t] = sum; __syncthreads();
    for (int o = 128; o > 0; o >>= 1) { if (t < o) red[t] += red[t + o]; __syncthreads(); }
    const float inv = 1.f / red[0];
    cnt = 0;
    for (int c = t; c < NTOK; c += 256, ++cnt) p[c] = vals[cnt] * inv;
}

// o = attn @ v; A-fragments converted f32->f16 on the fly (single pass over
// the 241MB attn tensor), B from v^T. Output scattered f16 into [B,N,C].
__global__ void gemm_attnv_kernel(const float* __restrict__ attn,
                                  const _Float16* __restrict__ vT16,
                                  _Float16* __restrict__ o16) {
    const int lane = threadIdx.x & 31;
    const int wave = (blockIdx.x * blockDim.x + threadIdx.x) >> 5;
    constexpr int TM = NTOK / 16;         // 140, nout = 64 = one wave tile
    if (wave >= TM * Bv * Hh) return;
    const int bh = wave / TM;
    const int m0 = (wave % TM) * 16;
    const int b = bh / Hh, hd = bh % Hh;
    const float* P = attn + (size_t)bh * NTOK * NTOK;
    const _Float16* Vt = vT16 + (size_t)bh * HD * NTOK;
    v8f acc[4] = {};
    const int ar = m0 + (lane & 15);
    for (int k0 = 0; k0 < NTOK; k0 += 32) {
        v16h a = load_a_f32(P, ar, NTOK, k0, lane);
#pragma unroll
        for (int s = 0; s < 4; ++s) {
            v16h bb = load_b(Vt, s * 16 + (lane & 15), NTOK, k0, lane);
            acc[s] = wmma_f16(a, bb, acc[s]);
        }
    }
    const int col = lane & 15;
    const int rbase = (lane >> 4) << 3;
#pragma unroll
    for (int s = 0; s < 4; ++s)
#pragma unroll
        for (int r = 0; r < 8; ++r) {
            const int q = m0 + rbase + r;
            const int d = s * 16 + col;
            o16[((size_t)(b * NTOK + q)) * Cv + hd * HD + d] = (_Float16)acc[s][r];
        }
}

// x1 += o @ w_proj^T + b_proj   (residual fused, f32 in place)
__global__ void gemm_proj_kernel(const _Float16* __restrict__ o16,
                                 const _Float16* __restrict__ w16,
                                 const float* __restrict__ bias,
                                 float* __restrict__ x1) {
    const int lane = threadIdx.x & 31;
    const int wave = (blockIdx.x * blockDim.x + threadIdx.x) >> 5;
    constexpr int TN = Cv / 64;           // 6
    constexpr int TM = Mv / 16;           // 280
    if (wave >= TM * TN) return;
    const int m0 = (wave / TN) * 16;
    const int n0 = (wave % TN) * 64;
    v8f acc[4] = {};
    gemm_nt_16x64(o16, Cv, w16, Cv, m0, n0, Cv, lane, acc);
    const int col = lane & 15;
    const int rbase = (lane >> 4) << 3;
#pragma unroll
    for (int s = 0; s < 4; ++s) {
        const int n = n0 + s * 16 + col;
#pragma unroll
        for (int r = 0; r < 8; ++r) {
            const size_t idx = (size_t)(m0 + rbase + r) * Cv + n;
            x1[idx] = x1[idx] + acc[s][r] + bias[n];
        }
    }
}

// hh = gelu_exact(ln2(x) @ w_fc1^T + b_fc1), f16 out
__global__ void gemm_fc1_kernel(const _Float16* __restrict__ h16,
                                const _Float16* __restrict__ w16,
                                const float* __restrict__ bias,
                                _Float16* __restrict__ hh16) {
    const int lane = threadIdx.x & 31;
    const int wave = (blockIdx.x * blockDim.x + threadIdx.x) >> 5;
    constexpr int TN = HID / 64;          // 24
    constexpr int TM = Mv / 16;           // 280
    if (wave >= TM * TN) return;
    const int m0 = (wave / TN) * 16;
    const int n0 = (wave % TN) * 64;
    v8f acc[4] = {};
    gemm_nt_16x64(h16, Cv, w16, Cv, m0, n0, Cv, lane, acc);
    const int col = lane & 15;
    const int rbase = (lane >> 4) << 3;
#pragma unroll
    for (int s = 0; s < 4; ++s) {
        const int n = n0 + s * 16 + col;
#pragma unroll
        for (int r = 0; r < 8; ++r) {
            const float u = acc[s][r] + bias[n];
            const float gel = 0.5f * u * (1.f + erff(u * 0.70710678118f));
            hh16[(size_t)(m0 + rbase + r) * HID + n] = (_Float16)gel;
        }
    }
}

// out_x = x1 + hh @ w_fc2^T + b_fc2  (final residual, f32 to d_out)
__global__ void gemm_fc2_kernel(const _Float16* __restrict__ hh16,
                                const _Float16* __restrict__ w16,
                                const float* __restrict__ bias,
                                const float* __restrict__ x1,
                                float* __restrict__ xout) {
    const int lane = threadIdx.x & 31;
    const int wave = (blockIdx.x * blockDim.x + threadIdx.x) >> 5;
    constexpr int TN = Cv / 64;           // 6
    constexpr int TM = Mv / 16;           // 280
    if (wave >= TM * TN) return;
    const int m0 = (wave / TN) * 16;
    const int n0 = (wave % TN) * 64;
    v8f acc[4] = {};
    gemm_nt_16x64(hh16, HID, w16, HID, m0, n0, HID, lane, acc);
    const int col = lane & 15;
    const int rbase = (lane >> 4) << 3;
#pragma unroll
    for (int s = 0; s < 4; ++s) {
        const int n = n0 + s * 16 + col;
#pragma unroll
        for (int r = 0; r < 8; ++r) {
            const size_t idx = (size_t)(m0 + rbase + r) * Cv + n;
            xout[idx] = x1[idx] + acc[s][r] + bias[n];
        }
    }
}

// ---------------------------------------------------------------- launcher

extern "C" void kernel_launch(void* const* d_in, const int* in_sizes, int n_in,
                              void* d_out, int out_size, void* d_ws, size_t ws_size,
                              hipStream_t stream) {
    const float* x      = (const float*)d_in[0];
    const float* pos    = (const float*)d_in[1];
    const float* ln1_g  = (const float*)d_in[2];
    const float* ln1_b  = (const float*)d_in[3];
    const float* w_qkv  = (const float*)d_in[4];
    const float* w_proj = (const float*)d_in[5];
    const float* b_proj = (const float*)d_in[6];
    const float* ln2_g  = (const float*)d_in[7];
    const float* ln2_b  = (const float*)d_in[8];
    const float* w_fc1  = (const float*)d_in[9];
    const float* b_fc1  = (const float*)d_in[10];
    const float* w_fc2  = (const float*)d_in[11];
    const float* b_fc2  = (const float*)d_in[12];

    float* out_x    = (float*)d_out;
    float* out_attn = out_x + (size_t)Mv * Cv;     // [B,H,N,N] f32

    char* ws = (char*)d_ws;
    auto alloc = [&](size_t bytes) {
        char* p = ws;
        ws += (bytes + 255) & ~(size_t)255;
        return p;
    };
    float*    x1    = (float*)   alloc((size_t)Mv * Cv * 4);
    _Float16* h16   = (_Float16*)alloc((size_t)Mv * Cv * 2);
    _Float16* q16   = (_Float16*)alloc((size_t)Bv * Hh * NTOK * HD * 2);
    _Float16* k16   = (_Float16*)alloc((size_t)Bv * Hh * NTOK * HD * 2);
    _Float16* vT16  = (_Float16*)alloc((size_t)Bv * Hh * NTOK * HD * 2);
    _Float16* o16   = (_Float16*)alloc((size_t)Mv * Cv * 2);
    _Float16* hh16  = (_Float16*)alloc((size_t)Mv * HID * 2);
    _Float16* wqkv16 = (_Float16*)alloc((size_t)TC * Cv * 2);
    _Float16* wproj16 = (_Float16*)alloc((size_t)Cv * Cv * 2);
    _Float16* wfc1_16 = (_Float16*)alloc((size_t)HID * Cv * 2);
    _Float16* wfc2_16 = (_Float16*)alloc((size_t)Cv * HID * 2);

    // weights -> f16 (once per call; tiny, L2-resident afterwards)
    auto cvt = [&](const float* s, _Float16* d, int n) {
        cvt_f32_to_f16_kernel<<<(n + 255) / 256, 256, 0, stream>>>(s, d, n);
    };
    cvt(w_qkv,  wqkv16,  TC * Cv);
    cvt(w_proj, wproj16, Cv * Cv);
    cvt(w_fc1,  wfc1_16, HID * Cv);
    cvt(w_fc2,  wfc2_16, Cv * HID);

    // x1 = x + pos ; h16 = ln1(x1)
    ln_kernel<<<Mv, 128, 0, stream>>>(x, pos, ln1_g, ln1_b, x1, h16);

    // qkv projection -> q (scaled) / k / v^T
    {
        const int waves = (Mv / 16) * (TC / 64);                 // 5040
        gemm_qkv_kernel<<<(waves * 32 + 255) / 256, 256, 0, stream>>>(
            h16, wqkv16, q16, k16, vT16);
    }
    // scores -> d_out attn region
    {
        const int waves = (NTOK / 16) * (NTOK / 64) * Bv * Hh;   // 58800
        gemm_scores_kernel<<<(waves * 32 + 255) / 256, 256, 0, stream>>>(
            q16, k16, out_attn);
    }
    // softmax in place
    softmax_kernel<<<Bv * Hh * NTOK, 256, 0, stream>>>(out_attn);

    // o = attn @ v
    {
        const int waves = (NTOK / 16) * Bv * Hh;                 // 1680
        gemm_attnv_kernel<<<(waves * 32 + 255) / 256, 256, 0, stream>>>(
            out_attn, vT16, o16);
    }
    // x1 += o @ w_proj^T + b_proj
    {
        const int waves = (Mv / 16) * (Cv / 64);                 // 1680
        gemm_proj_kernel<<<(waves * 32 + 255) / 256, 256, 0, stream>>>(
            o16, wproj16, b_proj, x1);
    }
    // h16 = ln2(x1)
    ln_kernel<<<Mv, 128, 0, stream>>>(x1, nullptr, ln2_g, ln2_b, nullptr, h16);

    // hh = gelu(h @ w_fc1^T + b_fc1)
    {
        const int waves = (Mv / 16) * (HID / 64);                // 6720
        gemm_fc1_kernel<<<(waves * 32 + 255) / 256, 256, 0, stream>>>(
            h16, wfc1_16, b_fc1, hh16);
    }
    // out_x = x1 + hh @ w_fc2^T + b_fc2
    {
        const int waves = (Mv / 16) * (Cv / 64);                 // 1680
        gemm_fc2_kernel<<<(waves * 32 + 255) / 256, 256, 0, stream>>>(
            hh16, wfc2_16, b_fc2, x1, out_x);
    }
}
